// VerticalRTColumnNet_28132035788937
// MI455X (gfx1250) — matrix operand order repaired and checked by reading, not compiled, not run on hardware
//
#include <hip/hip_runtime.h>
#include <hip/hip_bf16.h>
#include <math.h>

// ---------------------------------------------------------------------------
// Types for WMMA fragments (gfx1250, wave32)
// ---------------------------------------------------------------------------
typedef __attribute__((ext_vector_type(16))) _Float16     v16h;
typedef __attribute__((ext_vector_type(2)))  _Float16     v2h;
typedef __attribute__((ext_vector_type(8)))  float        v8f;
typedef __attribute__((ext_vector_type(4)))  unsigned int v4u;

union F16x16 { v4u q[2]; v16h h; };

// Compiler fence: kills CSE/hoisting of LDS data across pipeline stages so
// register liveness stays within one layer (keeps allocation < 256 VGPRs).
#define STAGE_FENCE() asm volatile("" ::: "memory")

static constexpr int Bn   = 16384;
static constexpr int Cc   = 6;
static constexpr int Ll   = 64;
static constexpr int OUTc = 4;
static constexpr int Ee   = 16;

// Staging: padded row stride so 16 rows spread across LDS banks (528 B row ->
// bank offset 4*row mod 64) instead of all rows aliasing the same 4 banks.
static constexpr int RS          = 264;      // halves per staged row (16B aligned)
static constexpr int BUF_HALVES  = 16 * RS;  // 4224 halves per tile buffer
static constexpr int WAVE_HALVES = 2 * BUF_HALVES;

// ---------------------------------------------------------------------------
// Weight packing: f32 (K x N row-major) -> f16 WMMA B-fragments.
// Fragment (kc, nt) is 1024 B: lane l stores 16 halves at lane*32 B.
// B-matrix (32x16 f16) layout: lanes 0-15 hold col = nt*16+l, K = kc*32 + {0..7, 16..23};
// lanes 16-31 hold the same cols, K = kc*32 + {8..15, 24..31}.  Frag id = nt*KC + kc.
// ---------------------------------------------------------------------------
__global__ void pack_w(const float* __restrict__ W, int K, int N, int KC, int NT,
                       _Float16* __restrict__ out) {
  int tid = blockIdx.x * blockDim.x + threadIdx.x;
  int total = KC * NT * 256;
  if (tid >= total) return;
  int frag = tid >> 8;
  int dw   = tid & 255;
  int lane = dw >> 3;
  int v    = dw & 7;
  int kc   = frag % KC;
  int nt   = frag / KC;
  int col  = nt * 16 + (lane & 15);
  int k0   = kc * 32 + ((lane < 16) ? 0 : 8) + 2 * v + ((v >= 4) ? 8 : 0);
  int k1   = k0 + 1;
  float a = (k0 < K && col < N) ? W[(size_t)k0 * N + col] : 0.0f;
  float b = (k1 < K && col < N) ? W[(size_t)k1 * N + col] : 0.0f;
  _Float16* dst = out + (size_t)frag * 512 + lane * 16 + v * 2;
  dst[0] = (_Float16)a;
  dst[1] = (_Float16)b;
}

// ---------------------------------------------------------------------------
// Device helpers
// ---------------------------------------------------------------------------
__device__ __forceinline__ v16h load_bfrag(const _Float16* __restrict__ Wp, int frag, int lane) {
  const v4u* p = (const v4u*)(Wp + (size_t)frag * 512 + lane * 16);
  F16x16 u;
  u.q[0] = p[0];
  u.q[1] = p[1];
  return u.h;
}

// WGP-scope prefetch (default scope = 0 -> pulls into all cache levels, unlike
// __builtin_prefetch which lowers to SCOPE_SE and only warms GL2, where the
// small weight tables already live).  Two 128 B lines from one address pair.
__device__ __forceinline__ void pf_2lines(const void* p) {
  unsigned long long a = (unsigned long long)p;
  asm volatile("global_prefetch_b8 %0, off\n\t"
               "global_prefetch_b8 %0, off offset:128" ::"v"(a));
}
__device__ __forceinline__ void pf_1line(const void* p) {
  unsigned long long a = (unsigned long long)p;
  asm volatile("global_prefetch_b8 %0, off" ::"v"(a));
}
// Prefetch an 8 KB fragment block (8 frags): 32 lanes x 256 B.
__device__ __forceinline__ void prefetch8k(const _Float16* p, int lane) {
  pf_2lines((const char*)p + lane * 256);
}
// Prefetch a single 1 KB fragment (enc1): 32 lanes x 32 B.
__device__ __forceinline__ void prefetch1k(const _Float16* p, int lane) {
  pf_1line((const char*)p + lane * 32);
}

// A-fragment (16x32 f16) loads from an LDS 16-row (stride RS) f16 buffer.
// lane<16: row=lane, halves {kc*32+0..7, kc*32+16..23};
// lane>=16: row=lane-16, halves {kc*32+8..15, kc*32+24..31}.
__device__ __forceinline__ void load_afrags(const _Float16* buf, int lane, v16h a[8]) {
  int r   = lane & 15;
  int off = (lane < 16) ? 0 : 8;
  const _Float16* row = buf + r * RS + off;
#pragma unroll
  for (int kc = 0; kc < 8; ++kc) {
    F16x16 u;
    u.q[0] = *(const v4u*)(row + kc * 32);
    u.q[1] = *(const v4u*)(row + kc * 32 + 16);
    a[kc] = u.h;
  }
}

__device__ __forceinline__ float sigm(float v) { return 1.0f / (1.0f + __expf(-v)); }

// C-matrix layout: lanes 0-15 -> rows 0..7 in vgprs 0..7, lanes 16-31 -> rows 8..15;
// col = lane & 15.  Convert f32->f16 first (cvt pairs pack into .l/.h of one
// dword), then ReLU as packed-f16 max: 8 cvt + 4 v_pk_max per tile.
__device__ __forceinline__ void store_relu16(v8f c, _Float16* outb, int nt, int lane) {
  int col = nt * 16 + (lane & 15);
  int rb  = (lane < 16) ? 0 : 8;
  v2h z;
  z[0] = (_Float16)0.0f;
  z[1] = (_Float16)0.0f;
#pragma unroll
  for (int v = 0; v < 8; v += 2) {
    v2h p;
    p[0] = (_Float16)c[v];
    p[1] = (_Float16)c[v + 1];
    p = __builtin_elementwise_max(p, z);
    outb[(rb + v) * RS + col]     = p[0];
    outb[(rb + v + 1) * RS + col] = p[1];
  }
}

// Full 256->256 layer with bias + ReLU, staged through LDS f16 buffers.
// Two independent accumulator chains (even/odd kc); nt loop kept rolled;
// B-fragments prefetched two iterations ahead (WGP scope); the epilogue
// (cvt/relu/ds_store) is software-pipelined one iteration behind so it
// executes in the shadow of the next iteration's loads instead of stalling
// in the WMMA->VALU hazard window.
__device__ __forceinline__ void layer256(const _Float16* inb, _Float16* outb,
                                         const _Float16* __restrict__ Wp,
                                         const float* biasL, int lane) {
  prefetch8k(Wp, lane);
  prefetch8k(Wp + 4096, lane);
  v16h a[8];
  load_afrags(inb, lane, a);
  v8f cprev;
#pragma unroll 1
  for (int nt = 0; nt < 16; ++nt) {
    prefetch8k(Wp + (nt + 2) * 4096, lane);  // depth-2 prefetch (tail overshoots harmlessly)
    v16h bf[8];
#pragma unroll
    for (int kc = 0; kc < 8; ++kc) bf[kc] = load_bfrag(Wp, nt * 8 + kc, lane);
    if (nt > 0) store_relu16(cprev, outb, nt - 1, lane);  // overlaps in-flight loads
    float bv = biasL[nt * 16 + (lane & 15)];
    v8f c0, c1;
#pragma unroll
    for (int v = 0; v < 8; ++v) { c0[v] = bv; c1[v] = 0.0f; }
#pragma unroll
    for (int kc = 0; kc < 8; kc += 2) {
      c0 = __builtin_amdgcn_wmma_f32_16x16x32_f16(false, a[kc], false, bf[kc], (short)0,
                                                  c0, false, false);
      c1 = __builtin_amdgcn_wmma_f32_16x16x32_f16(false, a[kc + 1], false, bf[kc + 1],
                                                  (short)0, c1, false, false);
    }
#pragma unroll
    for (int v = 0; v < 8; ++v) cprev[v] = c0[v] + c1[v];
  }
  store_relu16(cprev, outb, 15, lane);
}

// 256 -> 16 (only cols 0..3 valid) projection; returns C accumulator.
__device__ __forceinline__ v8f gemm16(const _Float16* inb,
                                      const _Float16* __restrict__ Wp,
                                      const float* biasL, int lane) {
  v16h a[8];
  load_afrags(inb, lane, a);
  float bv = ((lane & 15) < 4) ? biasL[lane & 15] : 0.0f;
  v8f c0, c1;
#pragma unroll
  for (int v = 0; v < 8; ++v) { c0[v] = bv; c1[v] = 0.0f; }
#pragma unroll
  for (int kc = 0; kc < 8; kc += 2) {
    v16h bf0 = load_bfrag(Wp, kc, lane);
    v16h bf1 = load_bfrag(Wp, kc + 1, lane);
    c0 = __builtin_amdgcn_wmma_f32_16x16x32_f16(false, a[kc], false, bf0, (short)0,
                                                c0, false, false);
    c1 = __builtin_amdgcn_wmma_f32_16x16x32_f16(false, a[kc + 1], false, bf1, (short)0,
                                                c1, false, false);
  }
#pragma unroll
  for (int v = 0; v < 8; ++v) c0[v] += c1[v];
  return c0;
}

// Store valid 4 columns of a C accumulator into an LDS [16][4] f32 block.
__device__ __forceinline__ void store_gate(v8f c, float* arr, int lane, bool sig) {
  int cl = lane & 15;
  if (cl < 4) {
    int rb = (lane < 16) ? 0 : 8;
#pragma unroll
    for (int v = 0; v < 8; ++v) {
      float val = c[v];
      if (sig) val = sigm(val);
      arr[(rb + v) * 4 + cl] = val;
    }
  }
}

// enc1 A-fragment built directly from x / emb (features: 6 x, 16 emb, 10 zero pad).
__device__ __forceinline__ v16h build_enc1(const float* __restrict__ x,
                                           const float* __restrict__ emb,
                                           int b0, int l, int lane) {
  int row = lane & 15;
  const float* xr = x + (size_t)(b0 + row) * (Cc * Ll) + l;
  const float* er = emb + l * Ee;
  float f[16];
  if (lane < 16) {  // K = 0..7, 16..23
    f[0] = xr[0];      f[1] = xr[64];    f[2] = xr[128];
    f[3] = xr[192];    f[4] = xr[256];   f[5] = xr[320];
    f[6] = er[0];      f[7] = er[1];
    f[8]  = er[10]; f[9]  = er[11]; f[10] = er[12];
    f[11] = er[13]; f[12] = er[14]; f[13] = er[15];
    f[14] = 0.0f; f[15] = 0.0f;
  } else {          // K = 8..15, 24..31
#pragma unroll
    for (int i = 0; i < 8; ++i) f[i] = er[2 + i];
#pragma unroll
    for (int i = 8; i < 16; ++i) f[i] = 0.0f;
  }
  v16h a;
#pragma unroll
  for (int i = 0; i < 16; ++i) a[i] = (_Float16)f[i];
  return a;
}

// ---------------------------------------------------------------------------
// Main fused kernel: one block = 16 batches x 64 levels.
// LDS layout: biases/small weights at offset 0 (DS immediates), then the hot
// staging buffers, then the rarely-touched gate arrays.
// ---------------------------------------------------------------------------
static constexpr size_t HEAD_FLOATS  = 256 * 4 + 32               // biases
                                       + 512 * 4 + 128 * 2 + 8 * 2  // sb/hd weights
                                       + 64;                        // u0
static constexpr size_t HEAD_BYTES   = HEAD_FLOATS * 4;           // 13760
static constexpr size_t STAGE_BYTES  = (size_t)8 * WAVE_HALVES * 2;  // 135168
static constexpr size_t GATES_BYTES  = (size_t)5 * 4096 * 4;      // 81920
static constexpr size_t SMEM_BYTES   = HEAD_BYTES + STAGE_BYTES + GATES_BYTES;

__global__ __launch_bounds__(256) void colnet_main(
    const float* __restrict__ x, const float* __restrict__ emb,
    const _Float16* __restrict__ w1p, const float* __restrict__ b1g,
    const _Float16* __restrict__ w2p, const float* __restrict__ b2g,
    const _Float16* __restrict__ td1p, const float* __restrict__ td1bg,
    const _Float16* __restrict__ td2p, const float* __restrict__ td2bg,
    const _Float16* __restrict__ sdp,  const float* __restrict__ sdbg,
    const _Float16* __restrict__ tu1p, const float* __restrict__ tu1bg,
    const _Float16* __restrict__ tu2p, const float* __restrict__ tu2bg,
    const _Float16* __restrict__ sup,  const float* __restrict__ subg,
    const _Float16* __restrict__ spp,  const float* __restrict__ spbg,
    const float* __restrict__ sbw1g, const float* __restrict__ sbb1g,
    const float* __restrict__ sbw2g, const float* __restrict__ sbb2g,
    const float* __restrict__ hdw1g, const float* __restrict__ hdb1g,
    const float* __restrict__ hdw2g, const float* __restrict__ hdb2g,
    float* __restrict__ out) {
  extern __shared__ char smem_raw[];
  float* b1L    = (float*)smem_raw;             // 256
  float* b2L    = b1L + 256;
  float* td1bL  = b2L + 256;
  float* tu1bL  = td1bL + 256;
  float* smallb = tu1bL + 256;                  // td2b sdb tu2b sub spb (pad 32)
  float* sbw1 = smallb + 32;
  float* sbb1 = sbw1 + 512;
  float* sbw2 = sbb1 + 128;
  float* sbb2 = sbw2 + 512;                     // 4 (pad 8)
  float* hdw1 = sbb2 + 8;
  float* hdb1 = hdw1 + 512;
  float* hdw2 = hdb1 + 128;
  float* hdb2 = hdw2 + 512;                     // 4 (pad 8)
  float* u0   = hdb2 + 8;                       // [16][4]
  _Float16* stage = (_Float16*)(smem_raw + HEAD_BYTES);
  float* Td = (float*)(smem_raw + HEAD_BYTES + STAGE_BYTES);  // [64][16][4]
  float* Sd = Td + 4096;                        // becomes D after down sweep
  float* Tu = Sd + 4096;
  float* Su = Tu + 4096;                        // becomes U after up sweep
  float* Sp = Su + 4096;

  const int tid  = threadIdx.x;
  const int lane = tid & 31;
  const int wave = tid >> 5;
  const int b0   = blockIdx.x * 16;

  // preload fp32 weights / biases into LDS (one pass, then barrier)
  for (int i = tid; i < 512; i += 256) {
    sbw1[i] = sbw1g[i]; sbw2[i] = sbw2g[i];
    hdw1[i] = hdw1g[i]; hdw2[i] = hdw2g[i];
  }
  if (tid < 128) { sbb1[tid] = sbb1g[tid]; hdb1[tid] = hdb1g[tid]; }
  if (tid < 4) { sbb2[tid] = sbb2g[tid]; hdb2[tid] = hdb2g[tid]; }
  {
    int i = tid;  // 256 threads cover 256 entries in one shot
    b1L[i] = b1g[i]; b2L[i] = b2g[i]; td1bL[i] = td1bg[i]; tu1bL[i] = tu1bg[i];
  }
  if (tid < 4) {
    smallb[tid]      = td2bg[tid];
    smallb[4 + tid]  = sdbg[tid];
    smallb[8 + tid]  = tu2bg[tid];
    smallb[12 + tid] = subg[tid];
    smallb[16 + tid] = spbg[tid];
  }
  __syncthreads();

  _Float16* hbuf = stage + (size_t)wave * WAVE_HALVES;
  _Float16* gbuf = hbuf + BUF_HALVES;

  // ---- per-level WMMA pipeline: each wave owns levels l = wave, wave+8, ... ----
#pragma unroll 1
  for (int l = wave; l < Ll; l += 8) {
    // encoder layer 1: (16 x 32) @ (32 x 256), epilogue pipelined by one nt
    prefetch1k(w1p, lane);
    prefetch1k(w1p + 512, lane);
    v16h a1 = build_enc1(x, emb, b0, l, lane);
    v8f cprev;
#pragma unroll 1
    for (int nt = 0; nt < 16; ++nt) {
      prefetch1k(w1p + (nt + 2) * 512, lane);
      v16h bf = load_bfrag(w1p, nt, lane);
      if (nt > 0) store_relu16(cprev, gbuf, nt - 1, lane);
      float bv = b1L[nt * 16 + (lane & 15)];
      v8f c;
#pragma unroll
      for (int v = 0; v < 8; ++v) c[v] = bv;
      cprev = __builtin_amdgcn_wmma_f32_16x16x32_f16(false, a1, false, bf, (short)0, c,
                                                     false, false);
    }
    store_relu16(cprev, gbuf, 15, lane);
    STAGE_FENCE();
    layer256(gbuf, hbuf, w2p, b2L, lane);       // h
    STAGE_FENCE();
    prefetch8k(td2p, lane);                     // warm projection table early
    layer256(hbuf, gbuf, td1p, td1bL, lane);    // relu(h @ td_w1 + b)
    STAGE_FENCE();
    { v8f c = gemm16(gbuf, td2p, smallb + 0, lane);  store_gate(c, Td + l * 64, lane, true); }
    STAGE_FENCE();
    prefetch8k(tu2p, lane);
    prefetch8k(sdp, lane);
    layer256(hbuf, gbuf, tu1p, tu1bL, lane);    // relu(h @ tu_w1 + b)
    STAGE_FENCE();
    { v8f c = gemm16(gbuf, tu2p, smallb + 8, lane);  store_gate(c, Tu + l * 64, lane, true); }
    STAGE_FENCE();
    prefetch8k(sup, lane);
    prefetch8k(spp, lane);
    { v8f c = gemm16(hbuf, sdp, smallb + 4, lane);   store_gate(c, Sd + l * 64, lane, false); }
    STAGE_FENCE();
    { v8f c = gemm16(hbuf, sup, smallb + 12, lane);  store_gate(c, Su + l * 64, lane, false); }
    STAGE_FENCE();
    { v8f c = gemm16(hbuf, spp, smallb + 16, lane);  store_gate(c, Sp + l * 64, lane, false); }
    STAGE_FENCE();
  }
  __syncthreads();

  // ---- down sweep: d_l = Td_l * d_{l-1} + Sd_l  (D stored into Sd) ----
  if (tid < 64) {
    int row = tid >> 2, j = tid & 3;
    float d = 0.0f;
    for (int l = 0; l < Ll; ++l) {
      int idx = l * 64 + row * 4 + j;
      d = Td[idx] * d + Sd[idx];
      Sd[idx] = d;
    }
  }
  __syncthreads();

  // ---- surface boundary: u0 = relu(D[L-1] @ sb_w1 + b) @ sb_w2 + b ----
  if (tid < 16) {
    int row = tid;
    float m0 = Sd[63 * 64 + row * 4 + 0];
    float m1 = Sd[63 * 64 + row * 4 + 1];
    float m2 = Sd[63 * 64 + row * 4 + 2];
    float m3 = Sd[63 * 64 + row * 4 + 3];
    float o0 = sbb2[0], o1 = sbb2[1], o2 = sbb2[2], o3 = sbb2[3];
    for (int k = 0; k < 128; ++k) {
      float a = sbb1[k] + m0 * sbw1[k] + m1 * sbw1[128 + k] + m2 * sbw1[256 + k] +
                m3 * sbw1[384 + k];
      a = fmaxf(a, 0.0f);
      o0 += a * sbw2[k * 4 + 0]; o1 += a * sbw2[k * 4 + 1];
      o2 += a * sbw2[k * 4 + 2]; o3 += a * sbw2[k * 4 + 3];
    }
    u0[row * 4 + 0] = o0; u0[row * 4 + 1] = o1;
    u0[row * 4 + 2] = o2; u0[row * 4 + 3] = o3;
  }
  __syncthreads();

  // ---- up sweep (reverse): u_l = Tu_l * u_{l+1} + Su_l  (U stored into Su) ----
  if (tid < 64) {
    int row = tid >> 2, j = tid & 3;
    float u = u0[row * 4 + j];
    for (int l = Ll - 1; l >= 0; --l) {
      int idx = l * 64 + row * 4 + j;
      u = Tu[idx] * u + Su[idx];
      Su[idx] = u;
    }
  }
  __syncthreads();

  // ---- head: out = relu((D+U+Sp) @ hd_w1 + b) @ hd_w2 + b ----
  for (int t = tid; t < 1024; t += 256) {
    int l = t >> 4, row = t & 15;
    int idx = l * 64 + row * 4;
    float m0 = Sd[idx + 0] + Su[idx + 0] + Sp[idx + 0];
    float m1 = Sd[idx + 1] + Su[idx + 1] + Sp[idx + 1];
    float m2 = Sd[idx + 2] + Su[idx + 2] + Sp[idx + 2];
    float m3 = Sd[idx + 3] + Su[idx + 3] + Sp[idx + 3];
    float o0 = hdb2[0], o1 = hdb2[1], o2 = hdb2[2], o3 = hdb2[3];
    for (int k = 0; k < 128; ++k) {
      float a = hdb1[k] + m0 * hdw1[k] + m1 * hdw1[128 + k] + m2 * hdw1[256 + k] +
                m3 * hdw1[384 + k];
      a = fmaxf(a, 0.0f);
      o0 += a * hdw2[k * 4 + 0]; o1 += a * hdw2[k * 4 + 1];
      o2 += a * hdw2[k * 4 + 2]; o3 += a * hdw2[k * 4 + 3];
    }
    size_t ob = (size_t)(b0 + row) * (OUTc * Ll) + l;
    out[ob + 0 * Ll] = o0; out[ob + 1 * Ll] = o1;
    out[ob + 2 * Ll] = o2; out[ob + 3 * Ll] = o3;
  }
}

// ---------------------------------------------------------------------------
// Host launcher
// ---------------------------------------------------------------------------
extern "C" void kernel_launch(void* const* d_in, const int* in_sizes, int n_in,
                              void* d_out, int out_size, void* d_ws, size_t ws_size,
                              hipStream_t stream) {
  const float* x      = (const float*)d_in[0];
  const float* emb    = (const float*)d_in[1];
  const float* enc_w1 = (const float*)d_in[2];
  const float* enc_b1 = (const float*)d_in[3];
  const float* enc_w2 = (const float*)d_in[4];
  const float* enc_b2 = (const float*)d_in[5];
  const float* td_w1  = (const float*)d_in[6];
  const float* td_b1  = (const float*)d_in[7];
  const float* td_w2  = (const float*)d_in[8];
  const float* td_b2  = (const float*)d_in[9];
  const float* sd_w   = (const float*)d_in[10];
  const float* sd_b   = (const float*)d_in[11];
  const float* sb_w1  = (const float*)d_in[12];
  const float* sb_b1  = (const float*)d_in[13];
  const float* sb_w2  = (const float*)d_in[14];
  const float* sb_b2  = (const float*)d_in[15];
  const float* tu_w1  = (const float*)d_in[16];
  const float* tu_b1  = (const float*)d_in[17];
  const float* tu_w2  = (const float*)d_in[18];
  const float* tu_b2  = (const float*)d_in[19];
  const float* su_w   = (const float*)d_in[20];
  const float* su_b   = (const float*)d_in[21];
  const float* sp_w   = (const float*)d_in[22];
  const float* sp_b   = (const float*)d_in[23];
  const float* hd_w1  = (const float*)d_in[24];
  const float* hd_b1  = (const float*)d_in[25];
  const float* hd_w2  = (const float*)d_in[26];
  const float* hd_b2  = (const float*)d_in[27];
  float* out = (float*)d_out;

  _Float16* ws = (_Float16*)d_ws;
  _Float16* w1p  = ws;                 // 16 frags
  _Float16* w2p  = w1p + 8192;         // 128 frags
  _Float16* td1p = w2p + 65536;
  _Float16* tu1p = td1p + 65536;
  _Float16* td2p = tu1p + 65536;       // 8 frags each below
  _Float16* sdp  = td2p + 4096;
  _Float16* tu2p = sdp + 4096;
  _Float16* sup  = tu2p + 4096;
  _Float16* spp  = sup + 4096;

  auto pack = [&](const float* W, int K, int N, int KC, int NT, _Float16* dst) {
    int total = KC * NT * 256;
    pack_w<<<(total + 255) / 256, 256, 0, stream>>>(W, K, N, KC, NT, dst);
  };
  pack(enc_w1, 22, 256, 1, 16, w1p);
  pack(enc_w2, 256, 256, 8, 16, w2p);
  pack(td_w1, 256, 256, 8, 16, td1p);
  pack(tu_w1, 256, 256, 8, 16, tu1p);
  pack(td_w2, 256, 4, 8, 1, td2p);
  pack(sd_w, 256, 4, 8, 1, sdp);
  pack(tu_w2, 256, 4, 8, 1, tu2p);
  pack(su_w, 256, 4, 8, 1, sup);
  pack(sp_w, 256, 4, 8, 1, spp);

  (void)hipFuncSetAttribute((const void*)colnet_main,
                            hipFuncAttributeMaxDynamicSharedMemorySize,
                            (int)SMEM_BYTES);

  colnet_main<<<Bn / 16, 256, SMEM_BYTES, stream>>>(
      x, emb, w1p, enc_b1, w2p, enc_b2, td1p, td_b1, td2p, td_b2, sdp, sd_b,
      tu1p, tu_b1, tu2p, tu_b2, sup, su_b, spp, sp_b,
      sb_w1, sb_b1, sb_w2, sb_b2, hd_w1, hd_b1, hd_w2, hd_b2, out);
}
